// MASAlignment_70643622085212
// MI455X (gfx1250) — matrix operand order repaired and checked by reading, not compiled, not run on hardware
//
#include <hip/hip_runtime.h>
#include <math.h>

// ---------------- problem constants ----------------
constexpr int NB   = 32;    // batch
constexpr int TXC  = 384;   // Tx
constexpr int TYC  = 1536;  // Ty
constexpr int DIMC = 256;   // feature dim of x
constexpr int NFC  = 80;    // NFEATS
constexpr int NFP  = 96;    // NFEATS padded to multiple of 32 (K for WMMA)
constexpr float NEGV    = -1000000000.0f;
constexpr float LPCONST = -73.51508265637381f; // -0.5*ln(2*pi)*80

// ---------------- workspace layout (bytes) ----------------
constexpr size_t OFF_YBF  = 0;                                   // bf16 [NB][TYC][NFP]  (feature-minor)
constexpr size_t SZ_YBF   = (size_t)NB * TYC * NFP * 2;
constexpr size_t OFF_XPBF = OFF_YBF + SZ_YBF;                    // bf16 [NB][TXC][NFP]  (feature-minor)
constexpr size_t SZ_XPBF  = (size_t)NB * TXC * NFP * 2;
constexpr size_t OFF_YSQ  = OFF_XPBF + SZ_XPBF;                  // f32 [NB][TYC]
constexpr size_t SZ_YSQ   = (size_t)NB * TYC * 4;
constexpr size_t OFF_XSQ  = OFF_YSQ + SZ_YSQ;                    // f32 [NB][TXC]
constexpr size_t SZ_XSQ   = (size_t)NB * TXC * 4;
constexpr size_t OFF_BITS = OFF_XSQ + SZ_XSQ;                    // u32 [NB][TYC][TXC/32]
constexpr size_t SZ_BITS  = (size_t)NB * TYC * (TXC / 32) * 4;
constexpr size_t OFF_WBF  = OFF_BITS + SZ_BITS;                  // bf16 [NFP][DIMC] zero-padded
constexpr size_t SZ_WBF   = (size_t)NFP * DIMC * 2;
constexpr size_t OFF_BPAD = OFF_WBF + SZ_WBF;                    // f32 [NFP] zero-padded bias
constexpr size_t SZ_BPAD  = (size_t)NFP * 4;
constexpr size_t OFF_LOSS = OFF_BPAD + SZ_BPAD;                  // f32 [1]

typedef __bf16 bf16_t;
typedef __bf16 v16bf  __attribute__((ext_vector_type(16)));
typedef __bf16 bf16x8 __attribute__((ext_vector_type(8)));
typedef float  v8f    __attribute__((ext_vector_type(8)));
typedef float  f32x4  __attribute__((ext_vector_type(4)));

// -------- K-1: pad W -> bf16 [NFP][DIMC], bias -> f32 [NFP] (no divergent GEMM guards)
__global__ void mas_prep_wb(const float* __restrict__ W, const float* __restrict__ bias,
                            bf16_t* __restrict__ Wbf, float* __restrict__ biasPad) {
  int tid = blockIdx.x * blockDim.x + threadIdx.x;
  if (tid >= NFP * DIMC) return;
  int f = tid / DIMC, d = tid - f * DIMC;
  Wbf[tid] = (f < NFC) ? (bf16_t)W[(size_t)f * DIMC + d] : (bf16_t)0.0f;
  if (tid < NFP) biasPad[tid] = (tid < NFC) ? bias[tid] : 0.0f;
}

// -------- K0: y -> bf16 [b][s][f] (padded), y_square, zero loss accum --------
__global__ void mas_prep_y(const float* __restrict__ y, bf16_t* __restrict__ ybfT,
                           float* __restrict__ ysq, float* __restrict__ lossAccum) {
  int tid = blockIdx.x * blockDim.x + threadIdx.x;
  if (tid == 0) *lossAccum = 0.0f;
  if (tid >= NB * TYC) return;
  int b = tid / TYC, s = tid - b * TYC;
  float acc = 0.0f;
  bf16x8* dst = (bf16x8*)(ybfT + ((size_t)b * TYC + s) * NFP);
  for (int fo = 0; fo < NFP / 8; ++fo) {
    bf16x8 pk;
#pragma unroll
    for (int i = 0; i < 8; ++i) {
      int f = fo * 8 + i;                       // uniform across lanes: no divergence
      float v = (f < NFC) ? y[((size_t)b * NFC + f) * TYC + s] : 0.0f;
      pk[i] = (bf16_t)v;
      acc += v * v;
    }
    dst[fo] = pk;
  }
  ysq[tid] = -0.5f * acc;
}

// -------- K1: xp = x*W^T + bias (WMMA bf16, f32 acc), store bf16 [b][t][f] ---
__global__ void mas_xp_gemm(const float* __restrict__ x, const bf16_t* __restrict__ Wbf,
                            const float* __restrict__ biasPad, bf16_t* __restrict__ xpT) {
  const int tileT = blockIdx.x;   // 0..23
  const int tileF = blockIdx.y;   // 0..5
  const int b     = blockIdx.z;
  const int lane  = threadIdx.x;  // one wave32
  const int m  = lane & 15;
  const int hi = lane >> 4;
  const int t0 = tileT * 16, f0 = tileF * 16;
  const int f  = f0 + m;          // B-fragment column (N) = lane&15
  const int kHalf = hi * 16;

  const float*  arow = x   + ((size_t)b * TXC + t0 + m) * DIMC + hi * 16;
  const bf16_t* brow = Wbf + (size_t)f * DIMC + kHalf;

  v8f acc = {};
#pragma unroll
  for (int kk = 0; kk < DIMC / 32; ++kk) {
    // A fragment: 16 contiguous floats per lane (k = 16*hi + s, matches B)
    f32x4 c0 = *(const f32x4*)(arow + kk * 32);
    f32x4 c1 = *(const f32x4*)(arow + kk * 32 + 4);
    f32x4 c2 = *(const f32x4*)(arow + kk * 32 + 8);
    f32x4 c3 = *(const f32x4*)(arow + kk * 32 + 12);
    v16bf a;
#pragma unroll
    for (int i = 0; i < 4; ++i) {
      a[i]      = (bf16_t)c0[i];
      a[4 + i]  = (bf16_t)c1[i];
      a[8 + i]  = (bf16_t)c2[i];
      a[12 + i] = (bf16_t)c3[i];
    }
    // B fragment: 16 contiguous bf16 in K-order = single 32B load
    v16bf bfr = *(const v16bf*)(brow + kk * 32);
    acc = __builtin_amdgcn_wmma_f32_16x16x32_bf16(false, a, false, bfr,
                                                  (short)0, acc, false, false);
  }
  float bv = biasPad[f];
#pragma unroll
  for (int r = 0; r < 8; ++r) {
    int t = t0 + r + hi * 8;                    // C/D layout: M = r (+8 hi half)
    xpT[((size_t)b * TXC + t) * NFP + f] = (bf16_t)(acc[r] + bv);
  }
}

// -------- K2: x_square from bf16 xp (contiguous per thread) -------------------
__global__ void mas_xsquare(const bf16_t* __restrict__ xpT, float* __restrict__ xsq) {
  int tid = blockIdx.x * blockDim.x + threadIdx.x;
  if (tid >= NB * TXC) return;
  int b = tid / TXC, t = tid - b * TXC;
  const bf16x8* row = (const bf16x8*)(xpT + ((size_t)b * TXC + t) * NFP);
  float acc = 0.0f;
  for (int fo = 0; fo < NFC / 8; ++fo) {
    bf16x8 v = row[fo];
#pragma unroll
    for (int i = 0; i < 8; ++i) { float w = (float)v[i]; acc += w * w; }
  }
  xsq[tid] = -0.5f * acc;
}

// -------- K3: cross-GEMM + log_prior, 4 N-tiles per wave, store [b][s][t] -----
// A[m][k]=y[b][k][s0+m] via [b][s][f] layout; B[k][n]=xp[b][k][t0+n] via [b][t][f].
constexpr int NT = 4;
__global__ void mas_cross_logprior(const bf16_t* __restrict__ ybfT,
                                   const bf16_t* __restrict__ xpT,
                                   const float* __restrict__ xsq,
                                   const float* __restrict__ ysq,
                                   const int* __restrict__ xlen,
                                   const int* __restrict__ ylen,
                                   float* __restrict__ lp) {
  const int tileT = blockIdx.x;   // 0..5 (64 t per block)
  const int tileS = blockIdx.y;   // 0..95
  const int b     = blockIdx.z;
  const int lane  = threadIdx.x;
  const int m  = lane & 15;
  const int hi = lane >> 4;
  const int s0 = tileS * 16, t0 = tileT * 64;
  const int kHalf = hi * 16;

  const bf16_t* arow  = ybfT + ((size_t)b * TYC + s0 + m) * NFP + kHalf;
  const bf16_t* brow0 = xpT  + ((size_t)b * TXC + t0 + m) * NFP + kHalf;

  v8f acc[NT] = {};
#pragma unroll
  for (int kk = 0; kk < NFP / 32; ++kk) {       // 3 K-steps (zero-padded to 96)
    bf16x8 alo = *(const bf16x8*)(arow + kk * 32);
    bf16x8 ahi = *(const bf16x8*)(arow + kk * 32 + 8);
    v16bf a;
#pragma unroll
    for (int i = 0; i < 8; ++i) { a[i] = alo[i]; a[8 + i] = ahi[i]; }
#pragma unroll
    for (int j = 0; j < NT; ++j) {
      v16bf bfr = *(const v16bf*)(brow0 + (size_t)j * 16 * NFP + kk * 32);
      acc[j] = __builtin_amdgcn_wmma_f32_16x16x32_bf16(false, a, false, bfr,
                                                       (short)0, acc[j], false, false);
    }
  }
  const int xl = xlen[b], yl = ylen[b];
#pragma unroll
  for (int j = 0; j < NT; ++j) {
    int t = t0 + j * 16 + m;
    float xs = xsq[b * TXC + t];
    bool tok = t < xl;
#pragma unroll
    for (int r = 0; r < 8; ++r) {
      int s = s0 + r + hi * 8;
      float val = acc[j][r] + xs + ysq[b * TYC + s] + LPCONST;
      // transposed layout lp[b][s][t]: coalesced rows for the DP scan
      lp[((size_t)b * TYC + s) * TXC + t] = (tok && s < yl) ? val : NEGV;
    }
  }
}

// -------- K4: forward DP; ping-pong Q rows => 1 barrier/step; ballot bits -----
__global__ void mas_dp_forward(const float* __restrict__ lp, unsigned* __restrict__ bits) {
  const int b = blockIdx.x;
  const int x = threadIdx.x;           // 0..383, 12 waves
  __shared__ float Q[2][TXC];
  Q[0][x] = (x == 0) ? 0.0f : NEGV;
  __syncthreads();
  const float* lpb = lp + (size_t)b * TYC * TXC;
  unsigned* bitb = bits + (size_t)b * TYC * (TXC / 32);
  int cur = 0;
  for (int y = 0; y < TYC; ++y) {
    float qc = Q[cur][x];
    float ql = (x > 0) ? Q[cur][x - 1] : NEGV;
    if (y > 0) {
      // bit[x] of row y-1: Q[y-1][x] < Q[y-1][x-1] (backtrack "move" test)
      unsigned msk = __builtin_amdgcn_ballot_w32((x > 0) && (qc < ql));
      if ((x & 31) == 0) bitb[(size_t)(y - 1) * (TXC / 32) + (x >> 5)] = msk;
    }
    float v = lpb[(size_t)y * TXC + x];
    if (y + 1 < TYC) __builtin_prefetch(&lpb[(size_t)(y + 1) * TXC + x], 0, 0);
    Q[cur ^ 1][x] = v + fmaxf(qc, ql);
    cur ^= 1;
    __syncthreads();
  }
}

// -------- K5: backtrack with cooperative LDS staging of decision bits ---------
// idx is monotone non-increasing, drops <=1 per y step => a 32-row window needs
// <=3 bit-words per row. Stage them into LDS so the serial walk never touches HBM.
__global__ void mas_backtrack(const unsigned* __restrict__ bits,
                              const int* __restrict__ xlen, const int* __restrict__ ylen,
                              const float* __restrict__ pred,
                              float* __restrict__ attn, int* __restrict__ durOut,
                              float* __restrict__ lossAccum) {
  const int b = blockIdx.x;
  const int x = threadIdx.x;  // 0..383
  __shared__ int      idxArr[TYC];
  __shared__ unsigned wbits[32 * 3];
  __shared__ int      sIdx;
  __shared__ float    lred[TXC];
  const int yl = ylen[b], xl = xlen[b];
  if (x == 0) sIdx = xl - 1;
  __syncthreads();
  const unsigned* bitb = bits + (size_t)b * TYC * (TXC / 32);

  for (int yb = TYC; yb > 0; yb -= 32) {       // rows y in [yb-32, yb)
    int idx0  = sIdx;
    int wHi   = idx0 >> 5;
    int wLo   = (idx0 > 32) ? ((idx0 - 32) >> 5) : 0;
    int ncols = wHi - wLo + 1;                 // <= 3
    if (x < 32 * ncols) {
      int rr = x / ncols, cc = x - rr * ncols;
      int yrow = yb - 33 + rr;                 // bits rows y-1 in [yb-33, yb-2]
      if (yrow >= 0) wbits[rr * 3 + cc] = bitb[(size_t)yrow * (TXC / 32) + wLo + cc];
    }
    __syncthreads();
    if (x == 0) {
      int idx = idx0;
      for (int y = yb - 1; y >= yb - 32; --y) {
        bool active = y < yl;
        idxArr[y] = active ? idx : -1;
        bool move = false;
        if (y > 0 && idx != 0 && active) {
          int rr = y - 1 - (yb - 33);
          unsigned w = wbits[rr * 3 + ((idx >> 5) - wLo)];
          move = (idx == y) || (((w >> (idx & 31)) & 1u) != 0);
        }
        idx -= move ? 1 : 0;
      }
      sIdx = idx;
    }
    __syncthreads();
  }

  // path (contiguous f32x4 per thread), durations, loss partials
  int dur = 0;
  f32x4* arow = (f32x4*)(attn + ((size_t)b * TXC + x) * TYC);
  for (int y4 = 0; y4 < TYC / 4; ++y4) {
    f32x4 out;
#pragma unroll
    for (int i = 0; i < 4; ++i) {
      int p = (idxArr[y4 * 4 + i] == x) ? 1 : 0;
      out[i] = (float)p;
      dur += p;
    }
    arow[y4] = out;
  }
  durOut[b * TXC + x] = dur;
  float tl = (x < xl) ? logf((float)dur + 1e-8f) : 0.0f;   // target_log * x_mask
  float d = pred[b * TXC + x] - tl;
  lred[x] = d * d;
  __syncthreads();
  if (x == 0) {
    float s = 0.0f;
    for (int i = 0; i < TXC; ++i) s += lred[i];
    atomicAdd(lossAccum, s);
  }
}

// -------- K6: finalize scalar loss ----------------
__global__ void mas_finalize(const float* __restrict__ lossAccum,
                             const int* __restrict__ xlen, float* __restrict__ lossOut) {
  if (blockIdx.x == 0 && threadIdx.x == 0) {
    int s = 0;
    for (int b = 0; b < NB; ++b) s += xlen[b];
    *lossOut = *lossAccum / (float)s;
  }
}

extern "C" void kernel_launch(void* const* d_in, const int* in_sizes, int n_in,
                              void* d_out, int out_size, void* d_ws, size_t ws_size,
                              hipStream_t stream) {
  (void)in_sizes; (void)n_in; (void)out_size; (void)ws_size;
  const float* x    = (const float*)d_in[0];
  const float* y    = (const float*)d_in[1];
  // d_in[2]=x_mask, d_in[3]=y_mask: recomputed from lengths
  const int*   xlen = (const int*)d_in[4];
  const int*   ylen = (const int*)d_in[5];
  const float* pred = (const float*)d_in[6];
  const float* W    = (const float*)d_in[7];
  const float* bias = (const float*)d_in[8];

  float* out     = (float*)d_out;
  float* attn    = out;                                  // [NB][TXC][TYC]
  float* lossOut = out + (size_t)NB * TXC * TYC;
  int*   durOut  = (int*)(out + (size_t)NB * TXC * TYC + 1);

  char* ws = (char*)d_ws;
  bf16_t*   ybfT      = (bf16_t*)(ws + OFF_YBF);
  bf16_t*   xpT       = (bf16_t*)(ws + OFF_XPBF);
  float*    ysq       = (float*)(ws + OFF_YSQ);
  float*    xsq       = (float*)(ws + OFF_XSQ);
  unsigned* bitsbuf   = (unsigned*)(ws + OFF_BITS);
  bf16_t*   Wbf       = (bf16_t*)(ws + OFF_WBF);
  float*    biasPad   = (float*)(ws + OFF_BPAD);
  float*    lossAccum = (float*)(ws + OFF_LOSS);

  mas_prep_wb<<<(NFP * DIMC + 255) / 256, 256, 0, stream>>>(W, bias, Wbf, biasPad);
  mas_prep_y<<<(NB * TYC + 255) / 256, 256, 0, stream>>>(y, ybfT, ysq, lossAccum);
  mas_xp_gemm<<<dim3(TXC / 16, NFP / 16, NB), 32, 0, stream>>>(x, Wbf, biasPad, xpT);
  mas_xsquare<<<(NB * TXC + 255) / 256, 256, 0, stream>>>(xpT, xsq);
  // log_prior scratch lives in the attn region of d_out (same size), layout [b][s][t]
  mas_cross_logprior<<<dim3(TXC / 64, TYC / 16, NB), 32, 0, stream>>>(
      ybfT, xpT, xsq, ysq, xlen, ylen, attn);
  mas_dp_forward<<<NB, TXC, 0, stream>>>(attn, bitsbuf);
  mas_backtrack<<<NB, TXC, 0, stream>>>(bitsbuf, xlen, ylen, pred, attn, durOut, lossAccum);
  mas_finalize<<<1, 1, 0, stream>>>(lossAccum, xlen, lossOut);
}
